// ProductionSwitchMoE_5325759447449
// MI455X (gfx1250) — matrix operand order repaired and checked by reading, not compiled, not run on hardware
//
#include <hip/hip_runtime.h>
#include <hip/hip_bf16.h>

// ---------------------------------------------------------------------------
// ProductionSwitchMoE for MI455X (gfx1250, wave32, WMMA)
//   T=8192 tokens, D=1024, H=4096, E=16 experts, capacity C=640
//   Memory-bound on fp32 expert weights; compute in bf16 WMMA.
//   FFN workgroup = (expert, 32-token block), H chunked 2x2048 so the
//   fused h-stage fits LDS; y accumulators persist in registers.
// ---------------------------------------------------------------------------

typedef __attribute__((ext_vector_type(16))) __bf16 v16bf;
typedef __attribute__((ext_vector_type(8)))  __bf16 v8bf;
typedef __attribute__((ext_vector_type(8)))  float  v8f;

#define T_TOK   8192
#define D_DIM   1024
#define H_DIM   4096
#define E_EXP   16
#define CAP     640
#define TB      32                    // tokens per workgroup (2 M-tiles)
#define NBLK    (CAP / TB)            // 20 token-blocks per expert
#define HC      2048                  // H chunk per LDS stage
#define XS_STR  1032                  // 1024 + 8 bf16 pad -> 4-DWORD bank skew
#define HS_STR  2056                  // 2048 + 8 bf16 pad -> 4-DWORD bank skew
#define SM_XS   0
#define SM_HS   (TB * XS_STR * 2)                 // 66048
#define SM_TOK  (SM_HS + TB * HS_STR * 2)         // 197632
#define SM_GATE (SM_TOK + 128)                    // 197760
#define SM_TOTAL (SM_GATE + 128)                  // 197888 bytes dynamic LDS

// Branch-free tanh-form GELU using hardware reciprocal:
//   t = 1 - 2*rcp(exp(u)+1) == tanh(u/2), safe at +-inf (rcp(inf)=0 -> t=1).
//   One v_exp_f32 + one v_rcp_f32 (both TRANS, co-issue with WMMA) + ~5 VALU.
__device__ __forceinline__ float gelu_f(float v) {
  float u  = 1.5957691216057308f * (v + 0.044715f * v * v * v);  // 2*sqrt(2/pi)*(...)
  float ex = __expf(u);
  float t  = 1.0f - 2.0f * __builtin_amdgcn_rcpf(ex + 1.0f);
  return 0.5f * v * (1.0f + t);
}

// ---------------------------------------------------------------------------
// Kernel 0: zero output + workspace counters/stats
// ---------------------------------------------------------------------------
__global__ __launch_bounds__(256) void moe_zero_k(float4* __restrict__ out4, int n4,
                                                  int* __restrict__ count,
                                                  float* __restrict__ dens,
                                                  float* __restrict__ psum,
                                                  float* __restrict__ zacc) {
  int i = blockIdx.x * 256 + threadIdx.x;
  if (i < n4) out4[i] = make_float4(0.f, 0.f, 0.f, 0.f);
  if (blockIdx.x == 0) {
    if (threadIdx.x < E_EXP) {
      count[threadIdx.x] = 0;
      dens[threadIdx.x]  = 0.f;
      psum[threadIdx.x]  = 0.f;
    }
    if (threadIdx.x == 0) zacc[0] = 0.f;
  }
}

// ---------------------------------------------------------------------------
// Kernel 1: router — logits, softmax, top-1 gate/index, aux-loss statistics
// ---------------------------------------------------------------------------
__global__ __launch_bounds__(256) void moe_router_k(const float* __restrict__ x,
                                                    const float* __restrict__ rw,
                                                    float* __restrict__ gates,
                                                    int*   __restrict__ eidx,
                                                    float* __restrict__ dens,
                                                    float* __restrict__ psum,
                                                    float* __restrict__ zacc) {
  __shared__ float s_ps[E_EXP];
  __shared__ float s_dn[E_EXP];
  __shared__ float s_z;
  if (threadIdx.x < E_EXP) { s_ps[threadIdx.x] = 0.f; s_dn[threadIdx.x] = 0.f; }
  if (threadIdx.x == 0) s_z = 0.f;
  __syncthreads();

  int t = blockIdx.x * 256 + threadIdx.x;
  float acc[E_EXP];
#pragma unroll
  for (int e = 0; e < E_EXP; ++e) acc[e] = 0.f;

  const float* xr = x + (size_t)t * D_DIM;
  for (int d = 0; d < D_DIM; d += 4) {
    float4 xv = *reinterpret_cast<const float4*>(xr + d);
#pragma unroll
    for (int e = 0; e < E_EXP; ++e) {
      float4 wv = *reinterpret_cast<const float4*>(rw + (size_t)e * D_DIM + d);
      acc[e] += xv.x * wv.x + xv.y * wv.y + xv.z * wv.z + xv.w * wv.w;
    }
  }
  // softmax, top-1
  float m = acc[0]; int am = 0;
#pragma unroll
  for (int e = 1; e < E_EXP; ++e) if (acc[e] > m) { m = acc[e]; am = e; }
  float p[E_EXP]; float s = 0.f;
#pragma unroll
  for (int e = 0; e < E_EXP; ++e) { p[e] = __expf(acc[e] - m); s += p[e]; }
  float inv = 1.f / s;            // max prob == exp(0)/s
  gates[t] = inv;
  eidx[t]  = am;

  // stats: prob mean, density counts, z-loss (logsumexp of probs, per reference)
  float se = 0.f;
#pragma unroll
  for (int e = 0; e < E_EXP; ++e) {
    float pr = p[e] * inv;
    atomicAdd(&s_ps[e], pr);
    se += __expf(pr);
  }
  atomicAdd(&s_dn[am], 1.0f);
  atomicAdd(&s_z, __logf(se));
  __syncthreads();
  if (threadIdx.x < E_EXP) {
    atomicAdd(&psum[threadIdx.x], s_ps[threadIdx.x]);
    atomicAdd(&dens[threadIdx.x], s_dn[threadIdx.x]);
  }
  if (threadIdx.x == 0) atomicAdd(zacc, s_z);
}

// ---------------------------------------------------------------------------
// Kernel 2: capacity-constrained dispatch (atomic append, clamp at CAP)
// ---------------------------------------------------------------------------
__global__ __launch_bounds__(256) void moe_dispatch_k(const int* __restrict__ eidx,
                                                      const float* __restrict__ gates,
                                                      int* __restrict__ count,
                                                      int* __restrict__ tok,
                                                      float* __restrict__ gsel) {
  int t = blockIdx.x * 256 + threadIdx.x;
  int e = eidx[t];
  int pos = atomicAdd(&count[e], 1);
  if (pos < CAP) {
    tok[e * CAP + pos]  = t;
    gsel[e * CAP + pos] = gates[t];
  }
}

// ---------------------------------------------------------------------------
// Kernel 3: expert FFN. Workgroup = (expert, 32-token block), 8 waves.
//   Per H-chunk of 2048:
//     GEMM1: h = gelu(x @ w1_chunk^T + b1)  -> 32 x 2048 bf16 in LDS
//     GEMM2: y += h @ w2_chunk^T            -> register accumulators
//   Each B fragment (fp32 weights -> bf16 in regs) feeds 2 WMMAs (2 M-tiles).
// ---------------------------------------------------------------------------
__global__ __launch_bounds__(256) void moe_ffn_k(const float* __restrict__ x,
                                                 const float* __restrict__ w1,
                                                 const float* __restrict__ b1,
                                                 const float* __restrict__ w2,
                                                 const float* __restrict__ b2,
                                                 const int* __restrict__ count,
                                                 const int* __restrict__ tok,
                                                 const float* __restrict__ gsel,
                                                 float* __restrict__ out) {
  extern __shared__ char smem[];
  __bf16* xs   = reinterpret_cast<__bf16*>(smem + SM_XS);
  __bf16* hs   = reinterpret_cast<__bf16*>(smem + SM_HS);
  int*    stok = reinterpret_cast<int*>(smem + SM_TOK);
  float*  sgat = reinterpret_cast<float*>(smem + SM_GATE);

  const int e   = blockIdx.x / NBLK;
  const int blk = blockIdx.x % NBLK;
  const int tid = threadIdx.x;

  int cnt = count[e];
  if (cnt > CAP) cnt = CAP;

  if (tid < TB) {
    int slot = blk * TB + tid;
    if (slot < cnt) {
      stok[tid] = tok[e * CAP + slot];
      sgat[tid] = gsel[e * CAP + slot];
    } else {
      stok[tid] = -1;
      sgat[tid] = 0.f;
    }
  }
  __syncthreads();

  // ---- stage gathered x rows into LDS as bf16 (coalesced global reads) ----
  for (int i = tid; i < TB * D_DIM; i += 256) {
    int r = i >> 10, c = i & (D_DIM - 1);
    int tk = stok[r];
    float v = (tk >= 0) ? x[(size_t)tk * D_DIM + c] : 0.f;
    xs[r * XS_STR + c] = (__bf16)v;
  }
  __syncthreads();

  const int wv   = tid >> 5;       // wave id 0..7
  const int lane = tid & 31;
  const int nlo  = lane & 15;      // column within N tile / A-row within M tile
  const int lhi  = lane >> 4;      // lane-group 0/1

  const float* w1e = w1 + (size_t)e * H_DIM * D_DIM;
  const float* b1e = b1 + (size_t)e * H_DIM;
  const float* w2e = w2 + (size_t)e * D_DIM * H_DIM;
  const float* b2e = b2 + (size_t)e * D_DIM;

  // persistent y accumulators: [m-tile][n-tile it], n-tile = (wv + 8*it)
  v8f yacc[2][8];
#pragma unroll
  for (int mt = 0; mt < 2; ++mt)
#pragma unroll
    for (int it = 0; it < 8; ++it) yacc[mt][it] = {};

  for (int hc = 0; hc < H_DIM / HC; ++hc) {
    // ---------------- GEMM1: [32 x D] x w1_chunk^T -> h chunk --------------
    for (int nt = wv; nt < HC / 16; nt += 8) {
      const int n_loc  = nt * 16 + nlo;
      const int n_glob = hc * HC + n_loc;
      const float* wr = w1e + (size_t)n_glob * D_DIM;
      v8f acc0 = {}, acc1 = {};
      for (int k0 = 0; k0 < D_DIM; k0 += 32) {
        const int ka = k0 + lhi * 8;
        // A fragments, M-tiles 0 and 1 (ISA 16-bit A 16x32 layout)
        v8bf a0l = *reinterpret_cast<const v8bf*>(xs + nlo * XS_STR + ka);
        v8bf a0h = *reinterpret_cast<const v8bf*>(xs + nlo * XS_STR + ka + 16);
        v8bf a1l = *reinterpret_cast<const v8bf*>(xs + (nlo + 16) * XS_STR + ka);
        v8bf a1h = *reinterpret_cast<const v8bf*>(xs + (nlo + 16) * XS_STR + ka + 16);
        v16bf a0, a1;
#pragma unroll
        for (int j = 0; j < 8; ++j) {
          a0[j] = a0l[j]; a0[j + 8] = a0h[j];
          a1[j] = a1l[j]; a1[j + 8] = a1h[j];
        }
        // B fragment: lane = column n, k at k0 + lhi*8 + {0..7, 16..23}
        const int kb = k0 + lhi * 8;
        __builtin_prefetch(wr + kb + 64, 0, 0);
        v16bf b;
#pragma unroll
        for (int j = 0; j < 16; j += 4) {
          float4 wv4 = *reinterpret_cast<const float4*>(wr + kb + j + (j & 8));
          b[j]     = (__bf16)wv4.x;
          b[j + 1] = (__bf16)wv4.y;
          b[j + 2] = (__bf16)wv4.z;
          b[j + 3] = (__bf16)wv4.w;
        }
        acc0 = __builtin_amdgcn_wmma_f32_16x16x32_bf16(false, a0, false, b,
                                                       (short)0, acc0, false, false);
        acc1 = __builtin_amdgcn_wmma_f32_16x16x32_bf16(false, a1, false, b,
                                                       (short)0, acc1, false, false);
      }
      const float bias = b1e[n_glob];
#pragma unroll
      for (int r = 0; r < 8; ++r) {
        hs[(r + 8 * lhi) * HS_STR + n_loc]        = (__bf16)gelu_f(acc0[r] + bias);
        hs[(r + 8 * lhi + 16) * HS_STR + n_loc]   = (__bf16)gelu_f(acc1[r] + bias);
      }
    }
    __syncthreads();

    // ---------------- GEMM2: y += h_chunk @ w2_chunk^T ---------------------
#pragma unroll
    for (int it = 0; it < 8; ++it) {
      const int n = (wv + 8 * it) * 16 + nlo;
      const float* wr = w2e + (size_t)n * H_DIM + (size_t)hc * HC;
      v8f acc0 = yacc[0][it], acc1 = yacc[1][it];
      for (int k0 = 0; k0 < HC; k0 += 32) {
        const int ka = k0 + lhi * 8;
        v8bf a0l = *reinterpret_cast<const v8bf*>(hs + nlo * HS_STR + ka);
        v8bf a0h = *reinterpret_cast<const v8bf*>(hs + nlo * HS_STR + ka + 16);
        v8bf a1l = *reinterpret_cast<const v8bf*>(hs + (nlo + 16) * HS_STR + ka);
        v8bf a1h = *reinterpret_cast<const v8bf*>(hs + (nlo + 16) * HS_STR + ka + 16);
        v16bf a0, a1;
#pragma unroll
        for (int j = 0; j < 8; ++j) {
          a0[j] = a0l[j]; a0[j + 8] = a0h[j];
          a1[j] = a1l[j]; a1[j + 8] = a1h[j];
        }
        const int kb = k0 + lhi * 8;
        __builtin_prefetch(wr + kb + 64, 0, 0);
        v16bf b;
#pragma unroll
        for (int j = 0; j < 16; j += 4) {
          float4 wv4 = *reinterpret_cast<const float4*>(wr + kb + j + (j & 8));
          b[j]     = (__bf16)wv4.x;
          b[j + 1] = (__bf16)wv4.y;
          b[j + 2] = (__bf16)wv4.z;
          b[j + 3] = (__bf16)wv4.w;
        }
        acc0 = __builtin_amdgcn_wmma_f32_16x16x32_bf16(false, a0, false, b,
                                                       (short)0, acc0, false, false);
        acc1 = __builtin_amdgcn_wmma_f32_16x16x32_bf16(false, a1, false, b,
                                                       (short)0, acc1, false, false);
      }
      yacc[0][it] = acc0;
      yacc[1][it] = acc1;
    }
    __syncthreads();   // before next chunk's GEMM1 overwrites hs
  }

  // ---------------- epilogue: bias, gate, predicated scatter ---------------
#pragma unroll
  for (int it = 0; it < 8; ++it) {
    const int n = (wv + 8 * it) * 16 + nlo;
    const float bias = b2e[n];
#pragma unroll
    for (int mt = 0; mt < 2; ++mt) {
#pragma unroll
      for (int r = 0; r < 8; ++r) {
        const int m  = mt * 16 + r + 8 * lhi;
        const int tk = stok[m];
        if (tk >= 0)
          out[(size_t)tk * D_DIM + n] = (yacc[mt][it][r] + bias) * sgat[m];
      }
    }
  }
}

// ---------------------------------------------------------------------------
// Kernel 4: aux loss finalize
// ---------------------------------------------------------------------------
__global__ void moe_aux_k(const float* __restrict__ dens,
                          const float* __restrict__ psum,
                          const float* __restrict__ zacc,
                          float* __restrict__ out) {
  if (blockIdx.x == 0 && threadIdx.x == 0) {
    const float invT = 1.0f / (float)T_TOK;
    float s = 0.f;
#pragma unroll
    for (int e = 0; e < E_EXP; ++e) s += (dens[e] * invT) * (psum[e] * invT);
    float aux = (float)E_EXP * s * 0.01f + (zacc[0] * invT) * 0.001f;
    out[(size_t)T_TOK * D_DIM] = aux;
  }
}

// ---------------------------------------------------------------------------
extern "C" void kernel_launch(void* const* d_in, const int* in_sizes, int n_in,
                              void* d_out, int out_size, void* d_ws, size_t ws_size,
                              hipStream_t stream) {
  const float* x  = (const float*)d_in[0];
  const float* rw = (const float*)d_in[1];
  const float* w1 = (const float*)d_in[2];
  const float* b1 = (const float*)d_in[3];
  const float* w2 = (const float*)d_in[4];
  const float* b2 = (const float*)d_in[5];
  float* out = (float*)d_out;

  char* ws = (char*)d_ws;
  float* gates = (float*)(ws);                         // T floats
  int*   eidx  = (int*)(ws + 32768);                   // T ints
  int*   count = (int*)(ws + 65536);                   // E ints
  float* dens  = (float*)(ws + 65536 + 64);            // E floats
  float* psum  = (float*)(ws + 65536 + 128);           // E floats
  float* zacc  = (float*)(ws + 65536 + 192);           // 1 float
  int*   tok   = (int*)(ws + 65536 + 256);             // E*CAP ints  (40960B)
  float* gsel  = (float*)(ws + 65536 + 256 + 40960);   // E*CAP floats (40960B)

  const int n4 = (T_TOK * D_DIM) / 4;
  moe_zero_k<<<(n4 + 255) / 256, 256, 0, stream>>>((float4*)out, n4, count, dens, psum, zacc);
  moe_router_k<<<T_TOK / 256, 256, 0, stream>>>(x, rw, gates, eidx, dens, psum, zacc);
  moe_dispatch_k<<<T_TOK / 256, 256, 0, stream>>>(eidx, gates, count, tok, gsel);
  moe_ffn_k<<<E_EXP * NBLK, 256, SM_TOTAL, stream>>>(x, w1, b1, w2, b2, count, tok, gsel, out);
  moe_aux_k<<<1, 32, 0, stream>>>(dens, psum, zacc, out);
}
